// GraphAttentionalLayer_39444979646678
// MI455X (gfx1250) — compile-verified, matrix operand order checked by reading, not run on hardware
//
#include <hip/hip_runtime.h>
#include <hip/hip_bf16.h>
#include <math.h>

#define NNODES 4096
#define FIN    512
#define NH     4
#define NHID   64
#define FOUT   256    // NH*NHID
#define BROWS  272    // 256 feature cols + 16-wide Z tile
#define NTILES 17     // BROWS/16
#define LDSROW 40     // padded LDS row length in f16 (32 data + 8 pad) -> bank-conflict-free

typedef __attribute__((ext_vector_type(16))) _Float16 v16h;
typedef __attribute__((ext_vector_type(8)))  float    v8f;

union AFragF { v16h v; _Float16 h[16]; };
union AFragU { v16h v; unsigned u[8]; };
union BFrag  { v16h v; uint4 q[2]; };

__device__ __forceinline__ v8f wmma_f32_16x16x32_f16(v16h a, v16h b, v8f c) {
  return __builtin_amdgcn_wmma_f32_16x16x32_f16(false, a, false, b, (short)0, c,
                                                false, false);
}

__device__ __forceinline__ unsigned pack01(int a, int b) {
  // two adjacency ints -> packed pair of f16 {0.0, 1.0}
  return ((a != 0) ? 0x3C00u : 0u) | ((b != 0) ? 0x3C000000u : 0u);
}

// ---------------------------------------------------------------- K0: W -> WT (f16, [n][k])
__global__ void k0_wt(const float* __restrict__ W, _Float16* __restrict__ WT) {
  int idx = blockIdx.x * 256 + threadIdx.x;   // 256*512 elements
  int n = idx >> 9;
  int k = idx & 511;
  WT[(size_t)n * FIN + k] = (_Float16)W[(size_t)k * FOUT + n];
}

// ---------------------------------------------------------------- K1: ht = leakyrelu(h @ W)
__global__ void __launch_bounds__(128) k1_gemm_ht(const float* __restrict__ h,
                                                  const _Float16* __restrict__ WT,
                                                  float* __restrict__ ht) {
  const int lane = threadIdx.x & 31;
  const int wid  = threadIdx.x >> 5;
  const int m    = lane & 15;
  const int half = lane >> 4;
  const int wave = blockIdx.x * 4 + wid;     // 4096 waves
  const int i0 = (wave >> 4) * 16;           // row tile
  const int n0 = (wave & 15) * 16;           // col tile

  const float*    hrow = h  + (size_t)(i0 + m) * FIN;
  const _Float16* wrow = WT + (size_t)(n0 + m) * FIN;

  v8f acc = {0.f, 0.f, 0.f, 0.f, 0.f, 0.f, 0.f, 0.f};
  for (int k0 = 0; k0 < FIN; k0 += 32) {
    AFragF A;
    const float* p0 = hrow + k0 + half * 8;
    const float* p1 = hrow + k0 + 16 + half * 8;
#pragma unroll
    for (int j = 0; j < 8; ++j) {
      A.h[j]     = (_Float16)p0[j];
      A.h[8 + j] = (_Float16)p1[j];
    }
    BFrag B;
    B.q[0] = *(const uint4*)(wrow + k0 + half * 8);
    B.q[1] = *(const uint4*)(wrow + k0 + 16 + half * 8);
    acc = wmma_f32_16x16x32_f16(A.v, B.v, acc);
  }
#pragma unroll
  for (int v = 0; v < 8; ++v) {
    float x = acc[v];
    x = (x > 0.f) ? x : 0.1f * x;            // leaky relu
    ht[(size_t)(i0 + half * 8 + v) * FOUT + (n0 + m)] = x;
  }
}

// ---------------------------------------------------------------- K2a: s_dst + per-head max
__global__ void k2a_scores(const float* __restrict__ ht, const float* __restrict__ a,
                           float* __restrict__ sdst, float* __restrict__ smax) {
  __shared__ float red[256];
  const int hd = blockIdx.x;                  // one block per head -> deterministic max
  const int t  = threadIdx.x;
  const float* ad = a + hd * 2 * NHID + NHID; // a_dst; a_src cancels in softmax
  float lmax = -INFINITY;
  for (int s = 0; s < NNODES / 256; ++s) {
    int j = t + 256 * s;
    const float* row = ht + (size_t)j * FOUT + hd * NHID;
    float acc = 0.f;
#pragma unroll
    for (int c = 0; c < NHID; ++c) acc += row[c] * ad[c];
    sdst[hd * NNODES + j] = acc;
    lmax = fmaxf(lmax, acc);
  }
  red[t] = lmax;
  __syncthreads();
  for (int off = 128; off > 0; off >>= 1) {
    if (t < off) red[t] = fmaxf(red[t], red[t + off]);
    __syncthreads();
  }
  if (t == 0) smax[hd] = red[0];
}

// ---------------------------------------------------------------- K2b: BT[n][j] = f16(E[j]*ht[j][n])
__global__ void k2b_scale_tr(const float* __restrict__ ht, const float* __restrict__ sdst,
                             const float* __restrict__ smax, _Float16* __restrict__ BT) {
  __shared__ float tile[32][33];
  const int j0 = blockIdx.x * 32;
  const int n0 = blockIdx.y * 32;
  const int tx = threadIdx.x, ty = threadIdx.y;
#pragma unroll
  for (int r = 0; r < 4; ++r) {
    int jj = ty + 8 * r;
    tile[jj][tx] = ht[(size_t)(j0 + jj) * FOUT + n0 + tx];
  }
  __syncthreads();
#pragma unroll
  for (int r = 0; r < 4; ++r) {
    int nn = ty + 8 * r;
    int n  = n0 + nn;
    int hd = n >> 6;
    int j  = j0 + tx;
    float E = __expf(sdst[hd * NNODES + j] - smax[hd]);
    BT[(size_t)n * NNODES + j] = (_Float16)(tile[tx][nn] * E);
  }
}

// ---------------------------------------------------------------- K2c: Z-tile rows (E columns + zeros)
__global__ void k2c_zrows(const float* __restrict__ sdst, const float* __restrict__ smax,
                          _Float16* __restrict__ BT) {
  int idx = blockIdx.x * 256 + threadIdx.x;  // 16*4096
  int r = idx >> 12;
  int j = idx & 4095;
  float v = 0.f;
  if (r < NH) v = __expf(sdst[r * NNODES + j] - smax[r]);
  BT[(size_t)(256 + r) * NNODES + j] = (_Float16)v;
}

// ---------------------------------------------------------------- K3: C = adj @ [E*ht | E]
// 4 waves/block share each 32-k slice of B via async global->LDS double buffering.
__global__ void __launch_bounds__(128) k3_attn(const int* __restrict__ adj,
                                               const _Float16* __restrict__ BT,
                                               float* __restrict__ C) {
  __shared__ _Float16 sB[2][NTILES * 16 * LDSROW];   // 2 x 10880 f16 = 43520 B

  const int tid  = threadIdx.x;
  const int lane = tid & 31;
  const int wid  = tid >> 5;
  const int m    = lane & 15;
  const int half = lane >> 4;
  const int rt = blockIdx.x * 4 + wid;       // 256 row tiles total
  const int i0 = rt * 16;
  const int* arow = adj + (size_t)(i0 + m) * NNODES;

  // async fill of one k-slice: BROWS rows x 32 k (64 B/row) -> LDS, padded rows
  auto fill = [&](int buf, int k0) {
    unsigned ldsbase = (unsigned)(uintptr_t)(&sB[buf][0]);
    for (int id = tid; id < BROWS * 4; id += 128) {
      int r = id >> 2, c = id & 3;                       // row, 16B chunk
      const _Float16* g = BT + (size_t)r * NNODES + k0 + c * 8;
      unsigned lds = ldsbase + (unsigned)(r * (LDSROW * 2) + c * 16);
      asm volatile("global_load_async_to_lds_b128 %0, %1, off"
                   :: "v"(lds), "v"(g) : "memory");
    }
  };

  const v8f vz = {0.f, 0.f, 0.f, 0.f, 0.f, 0.f, 0.f, 0.f};
  v8f acc[NTILES];
#pragma unroll
  for (int t = 0; t < NTILES; ++t) acc[t] = vz;

  fill(0, 0);                                            // prologue
  for (int kc = 0; kc < NNODES / 32; ++kc) {
    const int k0  = kc * 32;
    const int buf = kc & 1;
    asm volatile("s_wait_asynccnt 0x0" ::: "memory");    // own async stores to LDS done
    __syncthreads();                                     // all waves' slice visible
    if (kc + 1 < NNODES / 32) fill(buf ^ 1, k0 + 32);    // overlap next slice with compute

    if (k0 + 32 < NNODES) __builtin_prefetch(arow + k0 + 32, 0, 1);
    int4 q0 = *(const int4*)(arow + k0 + half * 8);
    int4 q1 = *(const int4*)(arow + k0 + half * 8 + 4);
    int4 q2 = *(const int4*)(arow + k0 + 16 + half * 8);
    int4 q3 = *(const int4*)(arow + k0 + 16 + half * 8 + 4);
    AFragU A;                                 // adj tile as f16 {0,1}, shared by all heads
    A.u[0] = pack01(q0.x, q0.y);  A.u[1] = pack01(q0.z, q0.w);
    A.u[2] = pack01(q1.x, q1.y);  A.u[3] = pack01(q1.z, q1.w);
    A.u[4] = pack01(q2.x, q2.y);  A.u[5] = pack01(q2.z, q2.w);
    A.u[6] = pack01(q3.x, q3.y);  A.u[7] = pack01(q3.z, q3.w);

#pragma unroll
    for (int t = 0; t < NTILES; ++t) {
      const int rb = (t * 16 + m) * LDSROW + half * 8;
      BFrag B;
      B.q[0] = *(const uint4*)&sB[buf][rb];
      B.q[1] = *(const uint4*)&sB[buf][rb + 16];
      acc[t] = wmma_f32_16x16x32_f16(A.v, B.v, acc[t]);
    }
  }
#pragma unroll
  for (int t = 0; t < NTILES; ++t)
#pragma unroll
    for (int v = 0; v < 8; ++v)
      C[(size_t)(i0 + half * 8 + v) * BROWS + t * 16 + m] = acc[t][v];
}

// ---------------------------------------------------------------- K4: out = C_features / C_Z
__global__ void k4_norm(const float* __restrict__ C, float* __restrict__ out) {
  int idx = blockIdx.x * 256 + threadIdx.x;  // 4096*256
  int i  = idx >> 8;
  int hc = idx & 255;
  int hd = hc >> 6;
  float z = C[(size_t)i * BROWS + 256 + hd];
  out[idx] = C[(size_t)i * BROWS + hc] / z;
}

extern "C" void kernel_launch(void* const* d_in, const int* in_sizes, int n_in,
                              void* d_out, int out_size, void* d_ws, size_t ws_size,
                              hipStream_t stream) {
  const float* h   = (const float*)d_in[0];
  const int*   adj = (const int*)d_in[1];
  const float* W   = (const float*)d_in[2];
  const float* a   = (const float*)d_in[3];
  float* out = (float*)d_out;

  char* ws = (char*)d_ws;
  // workspace layout (bytes)
  _Float16* WT   = (_Float16*)(ws);                              // 256*512*2      = 262144
  float*    htb  = (float*)(ws + 262144);                        // 4096*256*4     = 4194304
  float*    sdst = (float*)(ws + 4456448);                       // 4*4096*4       = 65536
  float*    smax = (float*)(ws + 4521984);                       // 256 reserved
  _Float16* BT   = (_Float16*)(ws + 4522240);                    // 272*4096*2     = 2228224
  float*    C    = (float*)(ws + 6750464);                       // 4096*272*4     = 4456448
  (void)in_sizes; (void)n_in; (void)out_size; (void)ws_size;

  k0_wt<<<512, 256, 0, stream>>>(W, WT);
  k1_gemm_ht<<<1024, 128, 0, stream>>>(h, WT, htb);
  k2a_scores<<<4, 256, 0, stream>>>(htb, a, sdst, smax);
  dim3 g2(NNODES / 32, FOUT / 32), b2(32, 8);
  k2b_scale_tr<<<g2, b2, 0, stream>>>(htb, sdst, smax, BT);
  k2c_zrows<<<256, 256, 0, stream>>>(sdst, smax, BT);
  k3_attn<<<64, 128, 0, stream>>>(adj, BT, C);
  k4_norm<<<4096, 256, 0, stream>>>(C, out);
}
